// SelectiveSSM_24739011625091
// MI455X (gfx1250) — matrix operand order repaired and checked by reading, not compiled, not run on hardware
//
#include <hip/hip_runtime.h>
#include <hip/hip_bf16.h>
#include <math.h>

// ---------------------------------------------------------------------------
// SelectiveSSM (Mamba-style block) for gfx1250 (MI455X).
// bf16 WMMA (v_wmma_f32_16x16x32_bf16) for all GEMMs; double-buffered TDM
// (tensor_load_to_lds) pipelines tile DMA against WMMA compute; fp32 VALU for
// layernorm / conv / softplus / scan / gating. Whole problem fits in L2.
// ---------------------------------------------------------------------------

typedef __attribute__((ext_vector_type(16))) __bf16 v16bf;
typedef __attribute__((ext_vector_type(8)))  float  v8f;
typedef __attribute__((ext_vector_type(4)))  int    v4i;
typedef __attribute__((ext_vector_type(8)))  int    v8i;

union BFrag { v16bf v; unsigned short u[16]; uint4 q[2]; };

__device__ __forceinline__ unsigned short f2bf(float f) {
    unsigned int u = __builtin_bit_cast(unsigned int, f);
    unsigned int r = u + 0x7FFFu + ((u >> 16) & 1u);   // round-to-nearest-even
    return (unsigned short)(r >> 16);
}

__device__ __forceinline__ float silu_f(float x) { return x / (1.0f + expf(-x)); }

// ---------------------------------------------------------------------------
// Tensor DMA descriptor builders (CDNA5 D#, CDNA5 ISA).
// Group0: count=1 | lds_addr | 57-bit global addr | type=2 ("image").
// ---------------------------------------------------------------------------
__device__ __forceinline__ v4i tdm_group0(unsigned lds_addr, unsigned long long gaddr) {
    v4i g;
    g.x = 1;                                            // count=1, user mode
    g.y = (int)lds_addr;                                // LDS byte address
    g.z = (int)(unsigned)(gaddr & 0xFFFFFFFFull);       // global_addr[31:0]
    g.w = (int)((((unsigned)(gaddr >> 32)) & 0x01FFFFFFu) | (2u << 30)); // [56:32] | type=2
    return g;
}

// Group1: 2D tile of 2-byte elements with LDS padding:
//   pad_enable=1, pad_interval=3 (every 16 DWORDs = 32 bf16 = one tile row),
//   pad_amount=3 (4 DWORDs = 8 shorts)  ->  LDS row stride 40 shorts.
__device__ __forceinline__ v8i tdm_group1(unsigned tdim0, unsigned tdim1,
                                          unsigned tiled0, unsigned tiled1,
                                          unsigned long long stride0) {
    v8i g;
    g.s0 = (1 << 16) | (1 << 20) | (3 << 22) | (3 << 25); // data_size=2B,pad cfg
    g.s1 = (int)((tdim0 & 0xFFFFu) << 16);                // tensor_dim0[15:0] @ bits 63:48
    g.s2 = (int)((tdim0 >> 16) | ((tdim1 & 0xFFFFu) << 16)); // dim0[31:16] | dim1[15:0]
    g.s3 = (int)((tdim1 >> 16) | (tiled0 << 16));         // dim1[31:16] | tile_dim0
    g.s4 = (int)(tiled1 & 0xFFFFu);                       // tile_dim1 | tile_dim2=0
    g.s5 = (int)(unsigned)(stride0 & 0xFFFFFFFFull);      // tensor_dim0_stride[31:0]
    g.s6 = (int)(unsigned)((stride0 >> 32) & 0xFFFFu);    // stride0[47:32] | stride1=0
    g.s7 = 0;
    return g;
}

__device__ __forceinline__ void tdm_load(v4i g0, v8i g1) {
    asm volatile("tensor_load_to_lds %0, %1" :: "s"(g0), "s"(g1) : "memory");
}

// ---------------------------------------------------------------------------
// Weight transpose+cast: src f32 [K][N] row-major -> dst bf16 [N][K]
// ---------------------------------------------------------------------------
__global__ __launch_bounds__(256)
void transcast_kernel(const float* __restrict__ src, unsigned short* __restrict__ dst,
                      int K, int N) {
    int idx = blockIdx.x * 256 + threadIdx.x;
    if (idx >= K * N) return;
    int n = idx / K;
    int k = idx - n * K;
    dst[idx] = f2bf(src[(size_t)k * N + n]);
}

// ---------------------------------------------------------------------------
// LayerNorm over D + cast to bf16. One block per row.
// ---------------------------------------------------------------------------
__global__ __launch_bounds__(256)
void ln_cast_kernel(const float* __restrict__ x, const float* __restrict__ g,
                    const float* __restrict__ b, unsigned short* __restrict__ xn,
                    int D) {
    int row = blockIdx.x;
    int tid = threadIdx.x;
    const float* xr = x + (size_t)row * D;
    float s1 = 0.f, s2 = 0.f;
    for (int i = tid; i < D; i += 256) { float v = xr[i]; s1 += v; s2 += v * v; }
    __shared__ float r1[256], r2[256];
    r1[tid] = s1; r2[tid] = s2;
    __syncthreads();
    for (int off = 128; off > 0; off >>= 1) {
        if (tid < off) { r1[tid] += r1[tid + off]; r2[tid] += r2[tid + off]; }
        __syncthreads();
    }
    float mu  = r1[0] / (float)D;
    float var = r2[0] / (float)D - mu * mu;
    float inv = rsqrtf(var + 1e-5f);
    for (int i = tid; i < D; i += 256) {
        float v = (xr[i] - mu) * inv * g[i] + b[i];
        xn[(size_t)row * D + i] = f2bf(v);
    }
}

// ---------------------------------------------------------------------------
// Generic bf16 WMMA GEMM:  C[M][N] = A[M][K](bf16) * Bt[N][K](bf16)^T
// Block tile 64x128, 8 waves, each wave 2x2 WMMA 16x16 tiles, K-step 32.
// Double-buffered TDM pipeline: wave 0 keeps one tensor_load_to_lds pair in
// flight (next K-tile) while all 8 waves run WMMA on the current tile;
// s_wait_tensorcnt<=2 retires the current buffer's DMA (TENSORcnt in-order).
// TDM hardware padding produces conflict-free 40-short LDS rows.
// mode 0: C = acc (+bias);  mode 1: C = softplus(acc + bias);
// mode 2: C = acc + res[row][col]
// ---------------------------------------------------------------------------
#define GBM 64
#define GBN 128
#define GBK 32
#define LPAD 8   // must match TDM pad: 4 DWORDs per 16-DWORD row

__global__ __launch_bounds__(256)
void gemm_bf16_kernel(const unsigned short* __restrict__ A,
                      const unsigned short* __restrict__ Bt,
                      const float* __restrict__ bias,
                      const float* __restrict__ res,
                      float* __restrict__ C,
                      int M, int N, int K, int mode) {
    __shared__ unsigned short lA[2][GBM][GBK + LPAD];
    __shared__ unsigned short lB[2][GBN][GBK + LPAD];

    const int tid  = threadIdx.x;
    const int bm   = blockIdx.y * GBM;
    const int bn   = blockIdx.x * GBN;
    const int w    = tid >> 5;
    const int lane = tid & 31;
    const int half = lane >> 4;
    const int mr   = lane & 15;
    const int m0   = (w >> 2) * 32;     // wave row offset within block (0/32)
    const int n0   = (w & 3) * 32;      // wave col offset within block (0..96)

    v8f acc[2][2];
    acc[0][0] = {}; acc[0][1] = {}; acc[1][0] = {}; acc[1][1] = {};

    const unsigned mRem = (unsigned)(M - bm) < (unsigned)GBM ? (unsigned)(M - bm) : (unsigned)GBM;
    const unsigned nRem = (unsigned)(N - bn) < (unsigned)GBN ? (unsigned)(N - bn) : (unsigned)GBN;
    const unsigned long long baseA = (unsigned long long)(size_t)A + 2ull * (size_t)bm * (size_t)K;
    const unsigned long long baseB = (unsigned long long)(size_t)Bt + 2ull * (size_t)bn * (size_t)K;
    const v8i g1A = tdm_group1(GBK, mRem, GBK, GBM, (unsigned long long)K);
    const v8i g1B = tdm_group1(GBK, nRem, GBK, GBN, (unsigned long long)K);

    const int nsteps = K / GBK;

    // Prologue: start DMA of K-tile 0 into buffer 0.
    if (w == 0) {
        tdm_load(tdm_group0((unsigned)(size_t)&lA[0][0][0], baseA), g1A);
        tdm_load(tdm_group0((unsigned)(size_t)&lB[0][0][0], baseB), g1B);
    }

    for (int s = 0; s < nsteps; ++s) {
        const int cur = s & 1;
        if (w == 0) {
            if (s + 1 < nsteps) {
                // Prefetch next K-tile into the other buffer, then wait until
                // only that transfer is outstanding (current tile landed).
                const unsigned long long koff = 2ull * (unsigned long long)(s + 1) * GBK;
                tdm_load(tdm_group0((unsigned)(size_t)&lA[cur ^ 1][0][0], baseA + koff), g1A);
                tdm_load(tdm_group0((unsigned)(size_t)&lB[cur ^ 1][0][0], baseB + koff), g1B);
                __builtin_amdgcn_s_wait_tensorcnt(2);   // 2 newest (prefetch pair) may remain
            } else {
                __builtin_amdgcn_s_wait_tensorcnt(0);
            }
        }
        __syncthreads();   // current buffer visible to all waves

        // A fragment layout (16-bit, 16x32): lanes 0-15 -> K{0..7,16..23},
        // lanes 16-31 -> K{8..15,24..31}.  Two b128 LDS reads per fragment.
        BFrag fa[2], fb[2];
        const int ac0 = half * 8;
        const int ac1 = 16 + half * 8;
#pragma unroll
        for (int i = 0; i < 2; ++i) {
            int rowA = m0 + i * 16 + mr;
            fa[i].q[0] = *(const uint4*)&lA[cur][rowA][ac0];
            fa[i].q[1] = *(const uint4*)&lA[cur][rowA][ac1];
        }
        // B fragment layout (16-bit, 32x16): lanes 0-15 -> K{0..7,16..23},
        // lanes 16-31 -> K{8..15,24..31}.
        const int bc0 = half * 8;
#pragma unroll
        for (int j = 0; j < 2; ++j) {
            int rowB = n0 + j * 16 + mr;
            fb[j].q[0] = *(const uint4*)&lB[cur][rowB][bc0];
            fb[j].q[1] = *(const uint4*)&lB[cur][rowB][bc0 + 16];
        }
#pragma unroll
        for (int i = 0; i < 2; ++i)
#pragma unroll
            for (int j = 0; j < 2; ++j)
                acc[i][j] = __builtin_amdgcn_wmma_f32_16x16x32_bf16(
                    false, fa[i].v, false, fb[j].v, (short)0, acc[i][j],
                    false, false);

        // Close the window before the next step's prefetch overwrites this buffer:
        // all frag reads issued+complete, then block-wide rendezvous.
        asm volatile("s_wait_dscnt 0x0" ::: "memory");
        __syncthreads();
    }

    // Epilogue. C/D layout: VGPR r -> row (r + half*8), lane mr -> col.
#pragma unroll
    for (int i = 0; i < 2; ++i) {
#pragma unroll
        for (int j = 0; j < 2; ++j) {
            int col = bn + n0 + j * 16 + mr;
#pragma unroll
            for (int r = 0; r < 8; ++r) {
                int row = bm + m0 + i * 16 + half * 8 + r;
                if (row < M && col < N) {
                    float v = acc[i][j][r];
                    if (mode == 1) {
                        v += bias[col];
                        v = (v > 20.f) ? v : log1pf(expf(v));   // softplus
                    } else {
                        if (bias) v += bias[col];
                        if (mode == 2) v += res[(size_t)row * N + col];
                    }
                    C[(size_t)row * N + col] = v;
                }
            }
        }
    }
}

// ---------------------------------------------------------------------------
// Causal depthwise conv (K=4) + bias + SiLU. Emits f32 and bf16 copies.
// xz layout: [B*T][2048]; x_ssm is columns 0..1023.
// ---------------------------------------------------------------------------
__global__ __launch_bounds__(256)
void conv_silu_kernel(const float* __restrict__ xz, const float* __restrict__ cw,
                      const float* __restrict__ cb, float* __restrict__ xs,
                      unsigned short* __restrict__ xsb, int Tlen) {
    int idx = blockIdx.x * 256 + threadIdx.x;        // over B*T*1024
    int d = idx & 1023;
    int t = (idx >> 10) % Tlen;
    int b = idx / (Tlen * 1024);
    float acc = cb[d];
#pragma unroll
    for (int k = 0; k < 4; ++k) {
        int ts = t - 3 + k;
        if (ts >= 0)
            acc += xz[((size_t)(b * Tlen + ts)) * 2048 + d] * cw[d * 4 + k];
    }
    float sv = silu_f(acc);
    xs[idx]  = sv;
    xsb[idx] = f2bf(sv);
}

// ---------------------------------------------------------------------------
// Sequential selective scan. One thread per channel, 16 states in registers.
// Fuses  y += x*D  and  y *= silu(z)  and bf16 cast for the output GEMM.
// ---------------------------------------------------------------------------
#define SC_THREADS 128
#define SC_CHUNK   64

__global__ __launch_bounds__(SC_THREADS)
void scan_kernel(const float* __restrict__ xs, const float* __restrict__ dtb,
                 const float* __restrict__ BC, const float* __restrict__ xz,
                 const float* __restrict__ log_A, const float* __restrict__ Dp,
                 unsigned short* __restrict__ yab, int Tlen) {
    const int b = blockIdx.y;
    const int d = blockIdx.x * SC_THREADS + threadIdx.x;   // channel, 0..1023
    const float Aa  = -expf(log_A[d]);
    const float Dpa = Dp[d];
    float h[16];
#pragma unroll
    for (int s = 0; s < 16; ++s) h[s] = 0.f;

    __shared__ float sB[SC_CHUNK][16];
    __shared__ float sC[SC_CHUNK][16];

    for (int t0 = 0; t0 < Tlen; t0 += SC_CHUNK) {
        __syncthreads();
        for (int i = threadIdx.x; i < SC_CHUNK * 32; i += SC_THREADS) {
            int tt = i >> 5, c = i & 31;
            float v = BC[((size_t)(b * Tlen + t0 + tt)) * 32 + c];
            if (c < 16) sB[tt][c] = v; else sC[tt][c - 16] = v;
        }
        __syncthreads();
        for (int tt = 0; tt < SC_CHUNK; ++tt) {
            const int t = t0 + tt;
            const size_t base = ((size_t)(b * Tlen + t)) * 1024 + d;
            const float xv  = xs[base];
            const float dtv = dtb[base];
            const float dec = expf(fminf(dtv * Aa, 0.f));
            const float dx  = dtv * xv;
            float y = 0.f;
#pragma unroll
            for (int s = 0; s < 16; ++s) {
                h[s] = h[s] * dec + dx * sB[tt][s];
                y += h[s] * sC[tt][s];
            }
            y += xv * Dpa;
            const float z = xz[((size_t)(b * Tlen + t)) * 2048 + 1024 + d];
            y *= silu_f(z);
            yab[base] = f2bf(y);
        }
    }
}

// ---------------------------------------------------------------------------
extern "C" void kernel_launch(void* const* d_in, const int* in_sizes, int n_in,
                              void* d_out, int out_size, void* d_ws, size_t ws_size,
                              hipStream_t stream) {
    (void)in_sizes; (void)n_in; (void)out_size; (void)ws_size;
    const int B = 2, T = 2048, D = 1024, DI = 1024;
    const int Rows = B * T;                       // 4096

    const float* x      = (const float*)d_in[0];
    const float* ln_g   = (const float*)d_in[1];
    const float* ln_b   = (const float*)d_in[2];
    const float* W_in   = (const float*)d_in[3];  // [1024][2048]
    const float* conv_w = (const float*)d_in[4];  // [1024][1][4]
    const float* conv_b = (const float*)d_in[5];
    const float* dt_w   = (const float*)d_in[6];  // [1024][1024]
    const float* dt_b   = (const float*)d_in[7];
    const float* B_w    = (const float*)d_in[8];  // [1024][16]
    const float* C_w    = (const float*)d_in[9];  // [1024][16]
    const float* log_A  = (const float*)d_in[10];
    const float* D_par  = (const float*)d_in[11];
    const float* W_out  = (const float*)d_in[12]; // [1024][1024]

    // ---- workspace layout (bytes), ~97 MB total -------------------------
    char* ws = (char*)d_ws;
    size_t off = 0;
    unsigned short* xn_bf  = (unsigned short*)(ws + off); off += (size_t)Rows * D * 2;        // 8 MB
    unsigned short* Wt_in  = (unsigned short*)(ws + off); off += (size_t)2048 * 1024 * 2;     // 4 MB
    unsigned short* Wt_dt  = (unsigned short*)(ws + off); off += (size_t)1024 * 1024 * 2;     // 2 MB
    unsigned short* Wt_out = (unsigned short*)(ws + off); off += (size_t)1024 * 1024 * 2;     // 2 MB
    unsigned short* BCt    = (unsigned short*)(ws + off); off += (size_t)32 * 1024 * 2;       // 64 KB
    float*          xz     = (float*)(ws + off);          off += (size_t)Rows * 2048 * 4;     // 32 MB
    float*          xs     = (float*)(ws + off);          off += (size_t)Rows * DI * 4;       // 16 MB
    unsigned short* xs_bf  = (unsigned short*)(ws + off); off += (size_t)Rows * DI * 2;       // 8 MB
    float*          dt     = (float*)(ws + off);          off += (size_t)Rows * DI * 4;       // 16 MB
    float*          BC     = (float*)(ws + off);          off += (size_t)Rows * 32 * 4;       // 512 KB
    unsigned short* ya_bf  = (unsigned short*)(ws + off); off += (size_t)Rows * DI * 2;       // 8 MB

    // ---- 1) weight transpose+cast to bf16 [N][K] ------------------------
    transcast_kernel<<<(2048 * 1024 + 255) / 256, 256, 0, stream>>>(W_in, Wt_in, 1024, 2048);
    transcast_kernel<<<(1024 * 1024 + 255) / 256, 256, 0, stream>>>(dt_w, Wt_dt, 1024, 1024);
    transcast_kernel<<<(1024 * 1024 + 255) / 256, 256, 0, stream>>>(W_out, Wt_out, 1024, 1024);
    transcast_kernel<<<(16 * 1024 + 255) / 256, 256, 0, stream>>>(B_w, BCt, 1024, 16);
    transcast_kernel<<<(16 * 1024 + 255) / 256, 256, 0, stream>>>(C_w, BCt + 16 * 1024, 1024, 16);

    // ---- 2) layernorm + cast --------------------------------------------
    ln_cast_kernel<<<Rows, 256, 0, stream>>>(x, ln_g, ln_b, xn_bf, D);

    // ---- 3) xz = xn @ W_in   [4096 x 2048] ------------------------------
    gemm_bf16_kernel<<<dim3(2048 / GBN, Rows / GBM), 256, 0, stream>>>(
        xn_bf, Wt_in, nullptr, nullptr, xz, Rows, 2048, 1024, 0);

    // ---- 4) depthwise causal conv + SiLU --------------------------------
    conv_silu_kernel<<<(Rows * DI) / 256, 256, 0, stream>>>(xz, conv_w, conv_b, xs, xs_bf, T);

    // ---- 5) dt = softplus(xs @ dt_w + dt_b) -----------------------------
    gemm_bf16_kernel<<<dim3(1024 / GBN, Rows / GBM), 256, 0, stream>>>(
        xs_bf, Wt_dt, dt_b, nullptr, dt, Rows, 1024, 1024, 1);

    // ---- 6) [Bm | Cm] = xs @ [B_w | C_w]   [4096 x 32] ------------------
    gemm_bf16_kernel<<<dim3(1, Rows / GBM), 256, 0, stream>>>(
        xs_bf, BCt, nullptr, nullptr, BC, Rows, 32, 1024, 0);

    // ---- 7) selective scan + x*D + silu(z) gating -----------------------
    scan_kernel<<<dim3(DI / SC_THREADS, B), SC_THREADS, 0, stream>>>(
        xs, dt, BC, xz, log_A, D_par, ya_bf, T);

    // ---- 8) out = residual + y @ W_out ----------------------------------
    gemm_bf16_kernel<<<dim3(1024 / GBN, Rows / GBM), 256, 0, stream>>>(
        ya_bf, Wt_out, nullptr, x, (float*)d_out, Rows, 1024, 1024, 2);
}
